// ModulatedConv2D_81063212744851
// MI455X (gfx1250) — matrix operand order repaired and checked by reading, not compiled, not run on hardware
//
#include <hip/hip_runtime.h>
#include <hip/hip_bf16.h>
#include <stdint.h>

// ---------------------------------------------------------------------------
// ModulatedConv2D (StyleGAN2), B=16, Cin=Cout=512, H=W=32, K=3, S=512
//
//   y[b,o] = d[b,o] * conv( style[b,:] * x[b,:], weight )[o]
//   d[b,o] = rsqrt( sum_i style[b,i]^2 * wsq[o,i] + 1e-8 )
//
// Conv = 16 batches of implicit GEMM  Y(512x1024) = sum_{t=0..8} Wt * Xshift.
// gfx1250 path: double-buffered global_load_async_to_lds_b128 staging
// (ASYNCcnt) -> ds_load_b128 fragments -> v_wmma_f32_16x16x32_bf16 (wave32).
// Activations are stored zero-padded (34x34) so the hot loop is branch-free.
// ---------------------------------------------------------------------------

typedef __attribute__((ext_vector_type(16))) __bf16 v16bf;
typedef __attribute__((ext_vector_type(8)))  float  v8f;
typedef unsigned int u32;
typedef u32 uint32x4 __attribute__((ext_vector_type(4)));

union Frag {                 // 32 bytes: 16 bf16 = 8 VGPRs (WMMA A/B operand)
    uint32x4 u[2];
    v16bf    v;
};

#define BATCH   16
#define CIN     512
#define COUT    512
#define HW      1024        // 32*32
#define SDIM    512
#define PW      34          // padded width/height
#define PPIX    (PW * PW)   // 1156 padded pixels

__device__ __forceinline__ unsigned short f32_to_bf16_rne(float f) {
    u32 u = __float_as_uint(f);
    u += 0x7FFFu + ((u >> 16) & 1u);        // round to nearest even
    return (unsigned short)(u >> 16);
}

// ---- style[b,c] = dot(w[b,:], style_w[c,:]) + style_b[c] ------------------
__global__ void style_kernel(const float* __restrict__ w,
                             const float* __restrict__ sw,
                             const float* __restrict__ sb,
                             float* __restrict__ style) {
    int idx = blockIdx.x * blockDim.x + threadIdx.x;   // 0..8191
    int b = idx >> 9, c = idx & 511;
    const float* wr  = w  + b * SDIM;
    const float* swr = sw + c * SDIM;
    float acc = 0.f;
    for (int s = 0; s < SDIM; ++s) acc = fmaf(wr[s], swr[s], acc);
    style[idx] = acc + sb[c];
}

// ---- wsq[o,i] = sum_t weight[o,i,t]^2 ; wbf[t][o][i] = bf16(weight) -------
__global__ void wprep_kernel(const float* __restrict__ weight,
                             float* __restrict__ wsq,
                             unsigned short* __restrict__ wbf) {
    int oi = blockIdx.x * blockDim.x + threadIdx.x;    // 0..262143
    const float* wp = weight + (size_t)oi * 9;
    float s = 0.f;
    #pragma unroll
    for (int t = 0; t < 9; ++t) {
        float v = wp[t];
        s = fmaf(v, v, s);
        wbf[(size_t)t * (COUT * CIN) + oi] = f32_to_bf16_rne(v);
    }
    wsq[oi] = s;
}

// ---- d[b,o] = rsqrt( sum_i style[b,i]^2 * wsq[o,i] + eps ) ----------------
__global__ void demod_kernel(const float* __restrict__ style,
                             const float* __restrict__ wsq,
                             float* __restrict__ dcoef) {
    int idx = blockIdx.x * blockDim.x + threadIdx.x;   // 0..8191
    int b = idx >> 9, o = idx & 511;
    const float* st = style + b * CIN;
    const float* wq = wsq   + o * CIN;
    float acc = 1e-8f;
    for (int i = 0; i < CIN; ++i) {
        float s = st[i];
        acc = fmaf(s * s, wq[i], acc);
    }
    dcoef[idx] = rsqrtf(acc);
}

// ---- clear padded activation buffer (borders must be zero) ----------------
__global__ void zero_kernel(uint32x4* __restrict__ p) {
    size_t i = (size_t)blockIdx.x * blockDim.x + threadIdx.x;
    p[i] = (uint32x4)0;
}

// ---- xTp[b][(h+1)*34+(w+1)][c] = bf16(style[b,c]*x[b,c,h,w]) (transpose) --
__global__ void xmod_kernel(const float* __restrict__ x,
                            const float* __restrict__ style,
                            unsigned short* __restrict__ xTp) {
    __shared__ float tile[32][33];
    int p0 = blockIdx.x * 32, c0 = blockIdx.y * 32, b = blockIdx.z;
    int tx = threadIdx.x, ty = threadIdx.y;            // (32, 8)
    #pragma unroll
    for (int r = 0; r < 32; r += 8) {
        int c = c0 + ty + r;
        float s = style[b * CIN + c];
        tile[ty + r][tx] = s * x[((size_t)b * CIN + c) * HW + p0 + tx];
    }
    __syncthreads();
    #pragma unroll
    for (int r = 0; r < 32; r += 8) {
        int p = p0 + ty + r;                           // interior pixel 0..1023
        int pp = ((p >> 5) + 1) * PW + (p & 31) + 1;   // padded pixel index
        xTp[((size_t)b * PPIX + pp) * CIN + c0 + tx] = f32_to_bf16_rne(tile[tx][ty + r]);
    }
}

// ---- main implicit-GEMM conv: async LDS staging + WMMA --------------------
// grid (8 Ntiles, 4 Mtiles, 16 batches), block 256 = 8 waves.
// Block tile 128x128; wave tile 64(M)x32(N) = 4x2 subtiles of 16x16.
// LDS: double buffer of (A: 128x32 bf16 = 8KB) + (B: 128pix x 32ch = 8KB).
__global__ __launch_bounds__(256) void modconv_wmma_kernel(
        const unsigned short* __restrict__ wbf,   // [9][512][512] bf16
        const unsigned short* __restrict__ xTp,   // [16][1156][512] bf16 (padded)
        const float* __restrict__ dcoef,          // [16][512]
        float* __restrict__ out) {                // [16][512][1024]
    __shared__ uint32x4 smem4[2048];              // 32 KB, 2 buffers x 16 KB

    const int tid   = threadIdx.x;
    const int lane  = tid & 31;
    const int wid   = tid >> 5;
    const int waveM = wid & 1;
    const int waveN = wid >> 1;
    const int b     = blockIdx.z;
    const int m_blk = blockIdx.y * 128;
    const int h0    = blockIdx.x * 4;             // first output row of N-tile
    const int lrow  = lane & 15;
    const int half  = lane >> 4;

    // ---- per-thread fill constants (loop-invariant) ----
    // A: 512 chunks of 16B (row 0..127, koff {0,8,16,24}); 2 chunks/thread.
    const int cA0 = tid, cA1 = tid + 256;
    const u32 voffA0 = (u32)(((cA0 >> 2) * CIN + (cA0 & 3) * 8) * 2);
    const u32 voffA1 = (u32)(((cA1 >> 2) * CIN + (cA1 & 3) * 8) * 2);
    const u32 lofA0  = (u32)(((cA0 >> 2) * 32 + (cA0 & 3) * 8) * 2);
    const u32 lofA1  = (u32)(((cA1 >> 2) * 32 + (cA1 & 3) * 8) * 2);
    // B: 512 chunks (local pixel 0..127, koff {0,8,16,24}); 2 chunks/thread.
    const int pB0 = cA0 >> 2, pB1 = cA1 >> 2;
    const u32 voffB0 = (u32)((((pB0 >> 5) * PW + (pB0 & 31)) * CIN + (cA0 & 3) * 8) * 2);
    const u32 voffB1 = (u32)((((pB1 >> 5) * PW + (pB1 & 31)) * CIN + (cA1 & 3) * 8) * 2);
    const u32 lofB0  = (u32)(8192 + ((pB0 * 32) + (cA0 & 3) * 8) * 2);
    const u32 lofB1  = (u32)(8192 + ((pB1 * 32) + (cA1 & 3) * 8) * 2);

    const u32 smem_base = (u32)(uintptr_t)&smem4[0];
    const unsigned short* xTpb = xTp + (size_t)b * PPIX * CIN;

    // async fill of one K-step tile (t = s>>4, k0 = (s&15)*32) into buffer buf
    auto fill = [&](int s, int buf) {
        const int t  = s >> 4;
        const int k0 = (s & 15) << 5;
        const int dh = t / 3 - 1;
        const int dw = t - (t / 3) * 3 - 1;
        const unsigned short* sA =
            wbf + (size_t)t * (COUT * CIN) + (size_t)m_blk * CIN + k0;
        const unsigned short* sB =
            xTpb + (size_t)((h0 + 1 + dh) * PW + 1 + dw) * CIN + k0;
        const u32 lA = smem_base + (u32)buf * 16384u;
        asm volatile("global_load_async_to_lds_b128 %0, %1, %2"
                     :: "v"(lA + lofA0), "v"(voffA0), "s"(sA) : "memory");
        asm volatile("global_load_async_to_lds_b128 %0, %1, %2"
                     :: "v"(lA + lofA1), "v"(voffA1), "s"(sA) : "memory");
        asm volatile("global_load_async_to_lds_b128 %0, %1, %2"
                     :: "v"(lA + lofB0), "v"(voffB0), "s"(sB) : "memory");
        asm volatile("global_load_async_to_lds_b128 %0, %1, %2"
                     :: "v"(lA + lofB1), "v"(voffB1), "s"(sB) : "memory");
    };

    // ---- per-lane fragment element indices into smem4 (16B units) ----
    // A frag (16x32 bf16): lane-half holds K {h*8..h*8+7} and {16+h*8..}
    int aIdx[4], bIdx[2];
    #pragma unroll
    for (int mi = 0; mi < 4; ++mi)
        aIdx[mi] = (waveM * 64 + mi * 16 + lrow) * 4 + half;          // +2 for hi
    // B frag (32x16 bf16): lane-half holds K {h*8..h*8+7} and {16+h*8..}
    #pragma unroll
    for (int ni = 0; ni < 2; ++ni)
        bIdx[ni] = 512 + (waveN * 32 + ni * 16 + lrow) * 4 + half;     // +2 for hi

    v8f acc[4][2] = {};

    fill(0, 0);

    #pragma unroll 2
    for (int s = 0; s < 144; ++s) {
        asm volatile("s_wait_asynccnt 0x0" ::: "memory");  // this wave's fills done
        __syncthreads();                                   // whole tile visible
        if (s + 1 < 144) fill(s + 1, (s + 1) & 1);         // overlap next fill

        const int base = (s & 1) * 1024;
        Frag af[4], bf[2];
        #pragma unroll
        for (int mi = 0; mi < 4; ++mi) {
            af[mi].u[0] = smem4[base + aIdx[mi]];
            af[mi].u[1] = smem4[base + aIdx[mi] + 2];
        }
        #pragma unroll
        for (int ni = 0; ni < 2; ++ni) {
            bf[ni].u[0] = smem4[base + bIdx[ni]];
            bf[ni].u[1] = smem4[base + bIdx[ni] + 2];
        }
        #pragma unroll
        for (int mi = 0; mi < 4; ++mi)
            #pragma unroll
            for (int ni = 0; ni < 2; ++ni)
                acc[mi][ni] = __builtin_amdgcn_wmma_f32_16x16x32_bf16(
                    false, af[mi].v, false, bf[ni].v,
                    (short)0, acc[mi][ni], false, false);
    }

    // store with demodulation; C/D layout: VGPR j -> M = half*8+j, N = lrow
    const float* db = dcoef + b * COUT;
    float* ob = out + (size_t)b * COUT * HW;
    const int n_wave = blockIdx.x * 128 + waveN * 32;
    #pragma unroll
    for (int mi = 0; mi < 4; ++mi) {
        int mrow0 = m_blk + waveM * 64 + mi * 16 + half * 8;
        #pragma unroll
        for (int ni = 0; ni < 2; ++ni) {
            int ncol = n_wave + ni * 16 + lrow;
            #pragma unroll
            for (int j = 0; j < 8; ++j) {
                int o = mrow0 + j;
                ob[(size_t)o * HW + ncol] = db[o] * acc[mi][ni][j];
            }
        }
    }
}

extern "C" void kernel_launch(void* const* d_in, const int* in_sizes, int n_in,
                              void* d_out, int out_size, void* d_ws, size_t ws_size,
                              hipStream_t stream) {
    const float* x       = (const float*)d_in[0];   // (16,512,32,32)
    const float* w       = (const float*)d_in[1];   // (16,512)
    const float* style_w = (const float*)d_in[2];   // (512,512)
    const float* style_b = (const float*)d_in[3];   // (512,)
    const float* weight  = (const float*)d_in[4];   // (512,512,3,3)
    float* out = (float*)d_out;                     // (16,512,32,32)

    // workspace carve (~25 MB), all 16B-aligned
    char* ws = (char*)d_ws;
    float* style = (float*)ws;                                  //  16*512 f32
    float* dcoef = style + BATCH * CIN;                         //  16*512 f32
    float* wsq   = dcoef + BATCH * COUT;                        // 512*512 f32
    unsigned short* wbf = (unsigned short*)(wsq + COUT * CIN);  // 9*512*512 bf16
    unsigned short* xTp = wbf + 9 * COUT * CIN;                 // 16*1156*512 bf16

    style_kernel<<<(BATCH * CIN) / 256, 256, 0, stream>>>(w, style_w, style_b, style);
    wprep_kernel<<<(COUT * CIN) / 256, 256, 0, stream>>>(weight, wsq, wbf);
    demod_kernel<<<(BATCH * COUT) / 256, 256, 0, stream>>>(style, wsq, dcoef);

    // zero padded activation buffer, then fill interior (transpose + modulate)
    const size_t xtp_vec4 = (size_t)BATCH * PPIX * CIN * 2 / 16;  // 16B units
    zero_kernel<<<(unsigned)(xtp_vec4 / 256), 256, 0, stream>>>((uint32x4*)xTp);
    dim3 tb(32, 8);
    dim3 tg(HW / 32, CIN / 32, BATCH);
    xmod_kernel<<<tg, tb, 0, stream>>>(x, style, xTp);

    dim3 gg(HW / 128, COUT / 128, BATCH);
    modconv_wmma_kernel<<<gg, 256, 0, stream>>>(wbf, xTp, dcoef, out);
}
